// MLP_87600152969580
// MI455X (gfx1250) — matrix operand order repaired and checked by reading, not compiled, not run on hardware
//
#include <hip/hip_runtime.h>
#include <hip/hip_bf16.h>
#include <stdint.h>
#include <stddef.h>

typedef __bf16 bf16_t;
typedef __attribute__((ext_vector_type(16))) __bf16 v16bf;
typedef __attribute__((ext_vector_type(8)))  float  v8f;
typedef __attribute__((ext_vector_type(4)))  float  f32x4;

#define MROWS  64
#define KP1    1440   /* 1424 padded to 45*32 */
#define KRAW   1424
#define N1     1024
#define N2     512
#define N3     256

/* LDS: region A = X[64][1440] bf16 (also reused as Y2[64][512] and logits),
        region B = Y1[64][1024] bf16 (also reused as Y3[64][256]). */
#define SMEM_X_BYTES  ((size_t)MROWS * KP1 * 2)
#define SMEM_Y_BYTES  ((size_t)MROWS * N1 * 2)
#define SMEM_BYTES    (SMEM_X_BYTES + SMEM_Y_BYTES)

struct Params {
  const int*    idx[16];
  const float*  emb[16];
  const bf16_t* W1T; const bf16_t* W2T; const bf16_t* W3T;
  const float*  b1;  const float*  b2;  const float*  b3;
  const float*  W4;  const float*  b4;
  float*        out;
};

/* widths (FACTOR*mult) and concat offsets of the 15 single-index features */
__constant__ int FW[15] = {128,128,128, 16, 16, 96,144, 64, 80, 64, 48,144, 96, 16,144};
__constant__ int FO[15] = {  0,128,256,384,400,416,512,656,720,800,848,912,1008,1152,1168};

/* Weight prep: W[K][N] f32 -> WT[N][Kpad] bf16 (zero-padded K). */
extern "C" __global__ void wprep_transpose_bf16(const float* __restrict__ W,
                                                bf16_t* __restrict__ WT,
                                                int K, int N, int Kpad)
{
  const long total = (long)N * Kpad;
  for (long i = (long)blockIdx.x * blockDim.x + threadIdx.x; i < total;
       i += (long)gridDim.x * blockDim.x) {
    const int n  = (int)(i / Kpad);
    const int kk = (int)(i % Kpad);
    const float v = (kk < K) ? W[(size_t)kk * N + n] : 0.f;
    WT[i] = (bf16_t)v;
  }
}

__device__ __forceinline__ void store_tile_relu(const v8f& acc, float bv,
                                                bf16_t* __restrict__ Out,
                                                int rowBase16, int half, int col,
                                                int ostride)
{
  bf16_t* orow = Out + (size_t)(rowBase16 + half * 8) * ostride + col;
  #pragma unroll
  for (int i = 0; i < 8; ++i) {
    float v = acc[i] + bv;
    v = v > 0.f ? v : 0.f;
    orow[(size_t)i * ostride] = (bf16_t)v;
  }
}

/* One dense layer: Out[64][nout] = relu(Ain[64][kpad] @ WT^T + bias), bf16 in/out,
   f32 accumulate via v_wmma_f32_16x16x32_bf16. 8 waves; each wave owns
   (nout/16/8) column blocks x 4 row blocks. */
__device__ __forceinline__ void gemm_relu_layer(
    const bf16_t* __restrict__ Ain, int kpad,
    const bf16_t* __restrict__ WT,  const float* __restrict__ bias,
    int nout, bf16_t* __restrict__ Out, int ostride)
{
  const int tid  = threadIdx.x;
  const int wave = tid >> 5;
  const int lane = tid & 31;
  const int half = lane >> 4;
  const int l16  = lane & 15;
  const int kIters    = kpad >> 5;
  const int cbPerWave = (nout >> 4) >> 3;

  /* A-fragment row pointers (ISA 16-bit A 16x32 layout: per lane, K runs
     [kb + half*8, +8) and [kb + 16 + half*8, +8) for row M = l16). */
  const bf16_t* a0 = Ain + (size_t)(l16 +  0) * kpad + half * 8;
  const bf16_t* a1 = Ain + (size_t)(l16 + 16) * kpad + half * 8;
  const bf16_t* a2 = Ain + (size_t)(l16 + 32) * kpad + half * 8;
  const bf16_t* a3 = Ain + (size_t)(l16 + 48) * kpad + half * 8;

  for (int cb = 0; cb < cbPerWave; ++cb) {
    const int n0  = (wave * cbPerWave + cb) << 4;
    const int col = n0 + l16;
    /* B-fragment (32x16 layout): lane holds K [kb + half*8, +8) and [kb + 16 + half*8, +8)
       for column N = l16; WT is [N][kpad] so this is fully coalesced. */
    const bf16_t* wrow = WT + (size_t)col * kpad + half * 8;
    __builtin_prefetch(wrow, 0, 1);   /* global_prefetch_b8 */

    v8f acc0 = {}, acc1 = {}, acc2 = {}, acc3 = {};
    for (int k = 0; k < kIters; ++k) {
      const int kb = k << 5;
      union { v16bf v; f32x4 q[2]; } B;
      B.q[0] = *(const f32x4*)(wrow + kb);
      B.q[1] = *(const f32x4*)(wrow + kb + 16);

      union { v16bf v; f32x4 q[2]; } A0, A1, A2, A3;
      A0.q[0] = *(const f32x4*)(a0 + kb);  A0.q[1] = *(const f32x4*)(a0 + kb + 16);
      A1.q[0] = *(const f32x4*)(a1 + kb);  A1.q[1] = *(const f32x4*)(a1 + kb + 16);
      A2.q[0] = *(const f32x4*)(a2 + kb);  A2.q[1] = *(const f32x4*)(a2 + kb + 16);
      A3.q[0] = *(const f32x4*)(a3 + kb);  A3.q[1] = *(const f32x4*)(a3 + kb + 16);

      acc0 = __builtin_amdgcn_wmma_f32_16x16x32_bf16(false, A0.v, false, B.v, (short)0, acc0, false, false);
      acc1 = __builtin_amdgcn_wmma_f32_16x16x32_bf16(false, A1.v, false, B.v, (short)0, acc1, false, false);
      acc2 = __builtin_amdgcn_wmma_f32_16x16x32_bf16(false, A2.v, false, B.v, (short)0, acc2, false, false);
      acc3 = __builtin_amdgcn_wmma_f32_16x16x32_bf16(false, A3.v, false, B.v, (short)0, acc3, false, false);
    }
    const float bv = bias[col];
    store_tile_relu(acc0, bv, Out,  0, half, col, ostride);
    store_tile_relu(acc1, bv, Out, 16, half, col, ostride);
    store_tile_relu(acc2, bv, Out, 32, half, col, ostride);
    store_tile_relu(acc3, bv, Out, 48, half, col, ostride);
  }
}

extern "C" __global__ void __launch_bounds__(256, 1)
fused_embed_mlp(Params p)
{
  extern __shared__ char smem[];
  bf16_t* X  = (bf16_t*)smem;                    /* [64][1440]; later Y2/logits */
  bf16_t* Y1 = (bf16_t*)(smem + SMEM_X_BYTES);   /* [64][1024]; later Y3 */
  const int tid     = threadIdx.x;
  const int rowBase = blockIdx.x * MROWS;

  /* ---- gather + concat + mean-pool into X (bf16), zero-pad K to 1440 ---- */
  {
    const int row  = tid >> 2;          /* 64 rows, 4 threads each */
    const int sub  = tid & 3;
    const int grow = rowBase + row;
    bf16_t* xr = X + (size_t)row * KP1;
    #pragma unroll
    for (int f = 0; f < 15; ++f) {
      const int w = FW[f], off = FO[f];
      const int id = p.idx[f][grow];
      const float* er = p.emb[f] + (size_t)id * w;
      for (int c = sub; c < w; c += 4)
        xr[off + c] = (bf16_t)er[c];
    }
    /* userids: mean over H=50, width 112, offset 1312 */
    const int*   uid = p.idx[15] + (size_t)grow * 50;
    const float* Eu  = p.emb[15];
    for (int c = sub; c < 112; c += 4) {
      float s = 0.f;
      for (int h = 0; h < 50; ++h)
        s += Eu[(size_t)uid[h] * 112 + c];
      xr[1312 + c] = (bf16_t)(s * 0.02f);
    }
    for (int c = KRAW + sub; c < KP1; c += 4)
      xr[c] = (bf16_t)0.f;
  }
  __syncthreads();

  gemm_relu_layer(X,  KP1, p.W1T, p.b1, N1, Y1, N1);   /* 1440 -> 1024 */
  __syncthreads();
  gemm_relu_layer(Y1, N1,  p.W2T, p.b2, N2, X,  N2);   /* 1024 -> 512 (Y2 aliases X) */
  __syncthreads();
  gemm_relu_layer(X,  N2,  p.W3T, p.b3, N3, Y1, N3);   /* 512 -> 256 (Y3 aliases Y1) */
  __syncthreads();

  /* ---- layer 4 (256 -> 2) + relu + softmax ---- */
  float* logits = (float*)X;
  if (tid < 128) {
    const int row = tid >> 1, col = tid & 1;
    const bf16_t* y3 = Y1 + (size_t)row * N3;
    float acc = p.b4[col];
    #pragma unroll 4
    for (int k = 0; k < N3; ++k)
      acc += (float)y3[k] * p.W4[2 * k + col];
    acc = acc > 0.f ? acc : 0.f;
    logits[2 * row + col] = acc;
  }
  __syncthreads();
  if (tid < MROWS) {
    const float l0 = logits[2 * tid], l1 = logits[2 * tid + 1];
    const float m  = l0 > l1 ? l0 : l1;
    const float e0 = __expf(l0 - m), e1 = __expf(l1 - m);
    const float inv = 1.f / (e0 + e1);
    float* o = p.out + (size_t)(rowBase + tid) * 2;
    o[0] = e0 * inv;
    o[1] = e1 * inv;
  }
}

extern "C" void kernel_launch(void* const* d_in, const int* in_sizes, int n_in,
                              void* d_out, int out_size, void* d_ws, size_t ws_size,
                              hipStream_t stream)
{
  (void)in_sizes; (void)n_in; (void)out_size; (void)ws_size;

  Params p;
  for (int f = 0; f < 16; ++f) {              /* dict order: idx_f, E_f interleaved */
    p.idx[f] = (const int*)  d_in[2 * f];
    p.emb[f] = (const float*)d_in[2 * f + 1];
  }
  const float* W1 = (const float*)d_in[32];  p.b1 = (const float*)d_in[33];
  const float* W2 = (const float*)d_in[34];  p.b2 = (const float*)d_in[35];
  const float* W3 = (const float*)d_in[36];  p.b3 = (const float*)d_in[37];
  p.W4  = (const float*)d_in[38];
  p.b4  = (const float*)d_in[39];
  p.out = (float*)d_out;

  /* workspace: transposed bf16 weights, K zero-padded */
  char* ws = (char*)d_ws;
  bf16_t* W1T = (bf16_t*)ws;                                          /* [1024][1440] */
  bf16_t* W2T = (bf16_t*)(ws + (size_t)N1 * KP1 * 2);                 /* [512][1024]  */
  bf16_t* W3T = (bf16_t*)(ws + (size_t)N1 * KP1 * 2 + (size_t)N2 * N1 * 2); /* [256][512] */
  p.W1T = W1T; p.W2T = W2T; p.W3T = W3T;

  wprep_transpose_bf16<<<512, 256, 0, stream>>>(W1, W1T, KRAW, N1, KP1);
  wprep_transpose_bf16<<<256, 256, 0, stream>>>(W2, W2T, N1,   N2, N1);
  wprep_transpose_bf16<<<128, 256, 0, stream>>>(W3, W3T, N2,   N3, N2);

  hipFuncSetAttribute((const void*)fused_embed_mlp,
                      hipFuncAttributeMaxDynamicSharedMemorySize, (int)SMEM_BYTES);
  fused_embed_mlp<<<65536 / MROWS, 256, SMEM_BYTES, stream>>>(p);
}
